// ClusteringLayer_58308476010591
// MI455X (gfx1250) — matrix operand order repaired and checked by reading, not compiled, not run on hardware
//
#include <hip/hip_runtime.h>

typedef __bf16 bf16_t;
typedef bf16_t v16bf __attribute__((ext_vector_type(16)));
typedef float  v8f   __attribute__((ext_vector_type(8)));

union Frag {
    v16bf v;
    uint4 q[2];
};

__device__ __forceinline__ unsigned pack_bf2(float a, float b) {
    unsigned ua = __float_as_uint(a);
    unsigned ub = __float_as_uint(b);
    ua = ua + 0x7FFFu + ((ua >> 16) & 1u);   // round-to-nearest-even
    ub = ub + 0x7FFFu + ((ub >> 16) & 1u);
    return (ua >> 16) | (ub & 0xFFFF0000u);
}

__device__ __forceinline__ v8f wmma_bf16(const Frag& a, const Frag& b, v8f c) {
    return __builtin_amdgcn_wmma_f32_16x16x32_bf16(
        false, a.v, false, b.v, (short)0, c, false, false);
}

// ---------------------------------------------------------------------------
// Prep: clusters f32 -> bf16 (u32-packed pairs, row-major [1024][256] u32)
//       + c_sq[1024]; one wave per cluster row. Runs once, ~2 MB read.
// ---------------------------------------------------------------------------
__global__ void __launch_bounds__(256)
prep_kernel(const float* __restrict__ clusters, unsigned* __restrict__ cbf,
            float* __restrict__ csq) {
    const int w    = threadIdx.x >> 5;
    const int lane = threadIdx.x & 31;
    const int row  = blockIdx.x * 8 + w;          // 0..1023

    const float* src = clusters + (size_t)row * 512;
    unsigned*    dst = cbf + (size_t)row * 256;

    float s = 0.f;
    #pragma unroll
    for (int j = 0; j < 8; ++j) {
        const float2 v = *(const float2*)(src + j * 64 + lane * 2);
        s += v.x * v.x + v.y * v.y;
        dst[j * 32 + lane] = pack_bf2(v.x, v.y);
    }
    s += __shfl_xor(s, 1, 32);
    s += __shfl_xor(s, 2, 32);
    s += __shfl_xor(s, 4, 32);
    s += __shfl_xor(s, 8, 32);
    s += __shfl_xor(s, 16, 32);
    if (lane == 0) csq[row] = s;
}

// ---------------------------------------------------------------------------
// Main: block = 32 rows x 1024 clusters, 256 threads (8 waves).
// Wave w owns clusters [w*128, +128) = 8 N-tiles x 2 M-tiles = 16 accums.
// B fragments double-buffered in registers (groups of 4 tiles) so global
// loads for the next group overlap the 8 WMMAs of the current group.
// ---------------------------------------------------------------------------
#define XS_STRIDE 260   // dwords per row (512 bf16 = 256 dwords + 4 pad)

__global__ void __launch_bounds__(256)
cluster_q_kernel(const float* __restrict__ x, const unsigned* __restrict__ cbf,
                 const float* __restrict__ csq, float* __restrict__ out) {
    __shared__ __align__(16) unsigned xs[32 * XS_STRIDE];   // 33.3 KB
    __shared__ float csq_s[1024];
    __shared__ float xsq_s[32];
    __shared__ float rowsum[32];

    const int tid  = threadIdx.x;
    const int lane = tid & 31;
    const int w    = tid >> 5;       // wave 0..7
    const int nlo  = lane & 15;      // tile-local N (and A-row) index
    const int g    = lane >> 4;      // lane half: selects K / M groups
    const int row0 = blockIdx.x * 32;

    for (int i = tid; i < 1024; i += 256) csq_s[i] = csq[i];
    if (tid < 32) rowsum[tid] = 0.f;

    // per-lane bases; all nt/ch offsets become compile-time immediates
    const unsigned* cb = cbf + ((size_t)(w * 128 + nlo) * 256 + g * 8);

    // ---- prime B group 0 (chunk 0, tiles 0..3): overlaps x staging ----
    Frag Bb[2][4];
    #pragma unroll
    for (int k = 0; k < 4; ++k) {
        const unsigned* bp = cb + k * 4096;
        Bb[0][k].q[0] = *(const uint4*)bp;
        Bb[0][k].q[1] = *(const uint4*)(bp + 4);
    }

    // ---- stage x tile (32 x 512) f32 -> bf16 in LDS; fuse ||x||^2 ----
    {
        const int r = tid >> 3;          // row 0..31 (8 contiguous lanes per row)
        const int c = tid & 7;
        const float* xrow = x + (size_t)(row0 + r) * 512;
        float s = 0.f;
        for (int slot = c; slot < 256; slot += 8) {
            const float2 v = *(const float2*)(xrow + slot * 2);
            s += v.x * v.x + v.y * v.y;
            xs[r * XS_STRIDE + slot] = pack_bf2(v.x, v.y);
        }
        s += __shfl_xor(s, 1, 32);       // reduce over the 8 loader lanes
        s += __shfl_xor(s, 2, 32);
        s += __shfl_xor(s, 4, 32);
        if ((lane & 7) == 0) xsq_s[r] = s;
    }
    __syncthreads();

    // ---- barrier-free GEMM loop: D = 512 in 16 chunks of K=32 ----
    v8f acc[16];
    #pragma unroll
    for (int t = 0; t < 16; ++t) acc[t] = (v8f)(0.f);

    const unsigned* xa0 = &xs[nlo * XS_STRIDE + g * 8];           // M-tile 0
    const unsigned* xa1 = &xs[(16 + nlo) * XS_STRIDE + g * 8];    // M-tile 1

    for (int ch = 0; ch < 16; ++ch) {
        Frag A0, A1;
        A0.q[0] = *(const uint4*)(xa0 + ch * 16);
        A0.q[1] = *(const uint4*)(xa0 + ch * 16 + 4);
        A1.q[0] = *(const uint4*)(xa1 + ch * 16);
        A1.q[1] = *(const uint4*)(xa1 + ch * 16 + 4);

        // prefetch B group 1 (tiles 4..7) of this chunk
        #pragma unroll
        for (int k = 0; k < 4; ++k) {
            const unsigned* bp = cb + (4 + k) * 4096 + ch * 16;
            Bb[1][k].q[0] = *(const uint4*)bp;
            Bb[1][k].q[1] = *(const uint4*)(bp + 4);
        }
        // compute on group 0 (tiles 0..3)
        #pragma unroll
        for (int k = 0; k < 4; ++k) {
            acc[k * 2 + 0] = wmma_bf16(A0, Bb[0][k], acc[k * 2 + 0]);
            acc[k * 2 + 1] = wmma_bf16(A1, Bb[0][k], acc[k * 2 + 1]);
        }
        // prefetch B group 0 (tiles 0..3) of next chunk (ch=15: rows <= 959,
        // still well inside the 1 MB bf16 buffer -> no OOB)
        #pragma unroll
        for (int k = 0; k < 4; ++k) {
            const unsigned* bp = cb + k * 4096 + (ch + 1) * 16;
            Bb[0][k].q[0] = *(const uint4*)bp;
            Bb[0][k].q[1] = *(const uint4*)(bp + 4);
        }
        // compute on group 1 (tiles 4..7)
        #pragma unroll
        for (int k = 0; k < 4; ++k) {
            acc[8 + k * 2 + 0] = wmma_bf16(A0, Bb[1][k], acc[8 + k * 2 + 0]);
            acc[8 + k * 2 + 1] = wmma_bf16(A1, Bb[1][k], acc[8 + k * 2 + 1]);
        }
    }
    __syncthreads();   // xsq_s / rowsum init visible to all

    // ---- epilogue: q = 1/(1+dist2), row sums, normalize, store ----
    float rs[16];
    #pragma unroll
    for (int i = 0; i < 16; ++i) rs[i] = 0.f;

    #pragma unroll
    for (int nt = 0; nt < 8; ++nt) {
        const int ncol = w * 128 + nt * 16 + nlo;
        const float cq = csq_s[ncol];
        #pragma unroll
        for (int mt = 0; mt < 2; ++mt) {
            #pragma unroll
            for (int r = 0; r < 8; ++r) {
                const int m = mt * 16 + r + 8 * g;          // local output row
                float d2 = xsq_s[m] + cq - 2.f * acc[nt * 2 + mt][r];
                d2 = fmaxf(d2, 0.f);
                const float qv = 1.f / (1.f + d2);          // ALPHA == 1
                acc[nt * 2 + mt][r] = qv;
                rs[mt * 8 + r] += qv;
            }
        }
    }
    #pragma unroll
    for (int i = 0; i < 16; ++i) {
        float s = rs[i];
        s += __shfl_xor(s, 1, 32);
        s += __shfl_xor(s, 2, 32);
        s += __shfl_xor(s, 4, 32);
        s += __shfl_xor(s, 8, 32);                          // 16-lane group sum
        if (nlo == 0) atomicAdd(&rowsum[(i >> 3) * 16 + (i & 7) + 8 * g], s);
    }
    __syncthreads();

    float inv[16];
    #pragma unroll
    for (int i = 0; i < 16; ++i)
        inv[i] = 1.f / rowsum[(i >> 3) * 16 + (i & 7) + 8 * g];

    #pragma unroll
    for (int nt = 0; nt < 8; ++nt) {
        const int ncol = w * 128 + nt * 16 + nlo;
        #pragma unroll
        for (int mt = 0; mt < 2; ++mt) {
            #pragma unroll
            for (int r = 0; r < 8; ++r) {
                const int m = mt * 16 + r + 8 * g;
                out[(size_t)(row0 + m) * 1024 + ncol] =
                    acc[nt * 2 + mt][r] * inv[mt * 8 + r];
            }
        }
    }
}

extern "C" void kernel_launch(void* const* d_in, const int* in_sizes, int n_in,
                              void* d_out, int out_size, void* d_ws, size_t ws_size,
                              hipStream_t stream) {
    const float* x        = (const float*)d_in[0];   // [65536, 512]
    const float* clusters = (const float*)d_in[1];   // [1024, 512]
    float* out = (float*)d_out;                      // [65536, 1024]

    unsigned* cbf = (unsigned*)d_ws;                           // 1 MB bf16 clusters
    float*    csq = (float*)((char*)d_ws + (1u << 20));        // 4 KB norms

    prep_kernel<<<dim3(128), dim3(256), 0, stream>>>(clusters, cbf, csq);
    cluster_q_kernel<<<dim3(65536 / 32), dim3(256), 0, stream>>>(x, cbf, csq, out);
}
